// MultiHeadSelfAttention_45380624450086
// MI455X (gfx1250) — matrix operand order repaired and checked by reading, not compiled, not run on hardware
//
#include <hip/hip_runtime.h>
#include <hip/hip_bf16.h>

typedef __attribute__((ext_vector_type(16))) _Float16 v16h;
typedef __attribute__((ext_vector_type(8)))  float    v8f;
typedef __attribute__((ext_vector_type(2)))  _Float16 h2;

#define WMMA_F16(a, b, c) \
  __builtin_amdgcn_wmma_f32_16x16x32_f16(false, (a), false, (b), (short)0, (c), false, false)

// Problem constants (B=16, T=1024, C=512, H=8, Dh=64)
constexpr int BB  = 16;
constexpr int TT  = 1024;
constexpr int CC  = 512;
constexpr int HH  = 8;
constexpr int DH  = 64;
constexpr int M1  = BB * TT;     // 16384 rows
constexpr int KT  = CC / 32;     // 16 k-tiles for both GEMMs (K = 512)

// ---------------------------------------------------------------------------
// Layout helpers (ISA 7.12.2, wave32):
// A 16x32 f16 fragment: lane = row + 16*((k&15)>=8); element e: v=e>>1,
//   k = (v<4?0:16) + 8*(lane>>4) + 2*(v&3) + (e&1)
// B 32x16 f16 fragment: lane = col + 16*((k&15)>=8); element e = 8*(k>=16) + (k&7)
// C 16x16 f32: VGPR r, lane l: M = r + 8*(l>>4), N = l&15
// Packed tiles are 512 halfs; lane's 16 halfs contiguous at lane*16.
// ---------------------------------------------------------------------------

__global__ __launch_bounds__(256) void pack_a_f16(const float* __restrict__ src,
                                                  _Float16* __restrict__ dst,
                                                  int K) {
  int idx = blockIdx.x * 256 + threadIdx.x;
  int tile   = idx >> 9;
  int within = idx & 511;
  int lane = within >> 4;
  int e    = within & 15;
  int ktiles = K >> 5;
  int tileM = tile / ktiles;
  int tileK = tile % ktiles;
  int row  = lane & 15;
  int half = lane >> 4;
  int v    = e >> 1;
  int kin  = ((v < 4) ? 0 : 16) + 8 * half + 2 * (v & 3) + (e & 1);
  int m = tileM * 16 + row;
  int k = tileK * 32 + kin;
  dst[idx] = (_Float16)src[(size_t)m * K + k];
}

__global__ __launch_bounds__(256) void pack_b_f16(const float* __restrict__ src,
                                                  _Float16* __restrict__ dst,
                                                  int K, int N) {
  int idx = blockIdx.x * 256 + threadIdx.x;
  int tile   = idx >> 9;
  int within = idx & 511;
  int lane = within >> 4;
  int e    = within & 15;
  int ktiles = K >> 5;
  int tileN = tile / ktiles;
  int tileK = tile % ktiles;
  int col   = lane & 15;
  int khalf = lane >> 4;
  int k = tileK * 32 + ((e < 8) ? (8 * khalf + e) : (16 + 8 * khalf + (e - 8)));
  int n = tileN * 16 + col;
  dst[idx] = (_Float16)src[(size_t)k * N + n];
}

// ---------------------------------------------------------------------------
// Wave mainloop: 32x64 output (2x4 WMMA tiles), K-loop over 32-wide steps.
// ---------------------------------------------------------------------------
__device__ __forceinline__ void gemm_wave_32x64(const _Float16* __restrict__ Ap,
                                                const _Float16* __restrict__ Bp,
                                                int ktiles, int tileM0, int tileN0,
                                                int lane, v8f acc[2][4]) {
  for (int kt = 0; kt < ktiles; ++kt) {
    v16h a0 = *(const v16h*)(Ap + (((size_t)(tileM0 + 0) * ktiles + kt) << 9) + lane * 16);
    v16h a1 = *(const v16h*)(Ap + (((size_t)(tileM0 + 1) * ktiles + kt) << 9) + lane * 16);
    v16h b0 = *(const v16h*)(Bp + (((size_t)(tileN0 + 0) * ktiles + kt) << 9) + lane * 16);
    v16h b1 = *(const v16h*)(Bp + (((size_t)(tileN0 + 1) * ktiles + kt) << 9) + lane * 16);
    v16h b2 = *(const v16h*)(Bp + (((size_t)(tileN0 + 2) * ktiles + kt) << 9) + lane * 16);
    v16h b3 = *(const v16h*)(Bp + (((size_t)(tileN0 + 3) * ktiles + kt) << 9) + lane * 16);
    acc[0][0] = WMMA_F16(a0, b0, acc[0][0]);
    acc[0][1] = WMMA_F16(a0, b1, acc[0][1]);
    acc[0][2] = WMMA_F16(a0, b2, acc[0][2]);
    acc[0][3] = WMMA_F16(a0, b3, acc[0][3]);
    acc[1][0] = WMMA_F16(a1, b0, acc[1][0]);
    acc[1][1] = WMMA_F16(a1, b1, acc[1][1]);
    acc[1][2] = WMMA_F16(a1, b2, acc[1][2]);
    acc[1][3] = WMMA_F16(a1, b3, acc[1][3]);
  }
}

// GEMM1: qkv = x @ Wqkv, epilogue scatters straight into packed Q/K/V
// fragment layouts (Q scaled by 1/sqrt(Dh)).
__global__ __launch_bounds__(256) void gemm_qkv(const _Float16* __restrict__ Ap,
                                                const _Float16* __restrict__ Bp,
                                                _Float16* __restrict__ Qp,
                                                _Float16* __restrict__ Kp,
                                                _Float16* __restrict__ Vp) {
  int lane = threadIdx.x & 31;
  int w    = threadIdx.x >> 5;
  int tileM0 = blockIdx.y * 8 + (w >> 1) * 2;
  int tileN0 = blockIdx.x * 8 + (w & 1) * 4;
  v8f acc[2][4];
#pragma unroll
  for (int i = 0; i < 2; ++i)
#pragma unroll
    for (int j = 0; j < 4; ++j)
#pragma unroll
      for (int r = 0; r < 8; ++r) acc[i][j][r] = 0.0f;

  gemm_wave_32x64(Ap, Bp, KT, tileM0, tileN0, lane, acc);

  int half = lane >> 4, col = lane & 15;
#pragma unroll
  for (int i = 0; i < 2; ++i)
#pragma unroll
    for (int j = 0; j < 4; ++j)
#pragma unroll
      for (int r = 0; r < 8; ++r) {
        float val = acc[i][j][r];
        int m = (tileM0 + i) * 16 + r + 8 * half;
        int n = (tileN0 + j) * 16 + col;
        int b = m >> 10, t = m & 1023;
        int which = n >> 9, h = (n >> 6) & 7, d = n & 63;
        int bh = b * HH + h;
        if (which == 0) {          // Q -> A-fragment pack, pre-scaled
          int qb = t >> 4, row = t & 15, kt = d >> 5, kk = d & 31;
          int lp = row + 16 * (((kk & 15) >= 8) ? 1 : 0);
          int v  = (kk < 16) ? ((kk & 7) >> 1) : (4 + ((kk & 7) >> 1));
          int e  = 2 * v + (kk & 1);
          Qp[((((size_t)bh * 64 + qb) * 2 + kt) << 9) + lp * 16 + e] =
              (_Float16)(val * 0.125f);
        } else if (which == 1) {   // K^T -> B-fragment pack
          int ktile = t >> 4, cc = t & 15, dc = d >> 5, kk = d & 31;
          int lp = cc + 16 * (((kk & 15) >= 8) ? 1 : 0);
          int e  = ((kk >= 16) ? 8 : 0) + (kk & 7);
          Kp[((((size_t)bh * 2 + dc) * 64 + ktile) << 9) + lp * 16 + e] = (_Float16)val;
        } else {                   // V -> B-fragment pack
          int kc = t >> 5, kk = t & 31, nt = d >> 4, cc = d & 15;
          int lp = cc + 16 * (((kk & 15) >= 8) ? 1 : 0);
          int e  = ((kk >= 16) ? 8 : 0) + (kk & 7);
          Vp[((((size_t)bh * 32 + kc) * 4 + nt) << 9) + lp * 16 + e] = (_Float16)val;
        }
      }
}

// ---------------------------------------------------------------------------
// Flash attention: one wave per (b,h, 16-query block); online softmax over
// 32-key chunks; P transposed C-layout -> A-layout via wave-private LDS.
// Writes output directly as f16 A-fragments for the final GEMM.
// ---------------------------------------------------------------------------
__global__ __launch_bounds__(256) void attn_flash(const _Float16* __restrict__ Qp,
                                                  const _Float16* __restrict__ Kp,
                                                  const _Float16* __restrict__ Vp,
                                                  _Float16* __restrict__ Yp) {
  __shared__ _Float16 sP[8][16][34];   // per-wave 16x32 P tile (+2 pad halfs)
  int lane = threadIdx.x & 31;
  int w    = threadIdx.x >> 5;
  int wid  = blockIdx.x * 8 + w;
  int qb   = wid & 63;          // query tile (16 rows)
  int bh   = wid >> 6;          // b*H + h
  int half = lane >> 4, col = lane & 15;

  v16h aq0 = *(const v16h*)(Qp + ((((size_t)bh * 64 + qb) * 2 + 0) << 9) + lane * 16);
  v16h aq1 = *(const v16h*)(Qp + ((((size_t)bh * 64 + qb) * 2 + 1) << 9) + lane * 16);

  float m8[8], l8[8];
  v8f o[4];
#pragma unroll
  for (int r = 0; r < 8; ++r) { m8[r] = -__builtin_inff(); l8[r] = 0.0f; }
#pragma unroll
  for (int nt = 0; nt < 4; ++nt)
#pragma unroll
    for (int r = 0; r < 8; ++r) o[nt][r] = 0.0f;

  v8f zf;
#pragma unroll
  for (int r = 0; r < 8; ++r) zf[r] = 0.0f;

  int jcount = (qb >> 1) + 1;
  for (int j = 0; j < jcount; ++j) {
    int kt0 = 2 * j, kt1 = 2 * j + 1;
    bool do1 = (kt1 <= qb);

    v16h bk00 = *(const v16h*)(Kp + ((((size_t)bh * 2 + 0) * 64 + kt0) << 9) + lane * 16);
    v16h bk10 = *(const v16h*)(Kp + ((((size_t)bh * 2 + 1) * 64 + kt0) << 9) + lane * 16);
    v8f s0 = WMMA_F16(aq0, bk00, zf);
    s0 = WMMA_F16(aq1, bk10, s0);

    v8f s1 = zf;
    if (do1) {
      v16h bk01 = *(const v16h*)(Kp + ((((size_t)bh * 2 + 0) * 64 + kt1) << 9) + lane * 16);
      v16h bk11 = *(const v16h*)(Kp + ((((size_t)bh * 2 + 1) * 64 + kt1) << 9) + lane * 16);
      s1 = WMMA_F16(aq0, bk01, zf);
      s1 = WMMA_F16(aq1, bk11, s1);
    }

    if (kt0 == qb) {               // diagonal tile: causal mask
#pragma unroll
      for (int r = 0; r < 8; ++r)
        s0[r] = (col > r + 8 * half) ? -__builtin_inff() : s0[r];
    }
    if (do1 && kt1 == qb) {
#pragma unroll
      for (int r = 0; r < 8; ++r)
        s1[r] = (col > r + 8 * half) ? -__builtin_inff() : s1[r];
    }

    float alpha[8], p0[8], p1[8];
#pragma unroll
    for (int r = 0; r < 8; ++r) {
      float t = do1 ? fmaxf(s0[r], s1[r]) : s0[r];
      t = fmaxf(t, __shfl_xor(t, 1, 32));
      t = fmaxf(t, __shfl_xor(t, 2, 32));
      t = fmaxf(t, __shfl_xor(t, 4, 32));
      t = fmaxf(t, __shfl_xor(t, 8, 32));
      float mn = fmaxf(m8[r], t);
      alpha[r] = __expf(m8[r] - mn);
      m8[r] = mn;
      p0[r] = __expf(s0[r] - mn);
      p1[r] = do1 ? __expf(s1[r] - mn) : 0.0f;
      float rs = p0[r] + p1[r];
      rs += __shfl_xor(rs, 1, 32);
      rs += __shfl_xor(rs, 2, 32);
      rs += __shfl_xor(rs, 4, 32);
      rs += __shfl_xor(rs, 8, 32);
      l8[r] = l8[r] * alpha[r] + rs;
    }

    // C-layout -> LDS (16 rows x 32 keys of this chunk)
#pragma unroll
    for (int r = 0; r < 8; ++r) {
      int row16 = r + 8 * half;
      sP[w][row16][col]      = (_Float16)p0[r];
      sP[w][row16][16 + col] = (_Float16)p1[r];
    }
    asm volatile("s_wait_dscnt 0" ::: "memory");

    // LDS -> A-fragment of P (lane holds row = lane&15)
    v16h pa;
#pragma unroll
    for (int v = 0; v < 8; ++v) {
      int k0 = ((v < 4) ? 0 : 16) + 8 * half + 2 * (v & 3);
      h2 pr = *(const h2*)&sP[w][col][k0];
      pa[2 * v]     = pr.x;
      pa[2 * v + 1] = pr.y;
    }

    // rescale running output and accumulate P @ V
#pragma unroll
    for (int nt = 0; nt < 4; ++nt) {
#pragma unroll
      for (int r = 0; r < 8; ++r) o[nt][r] *= alpha[r];
      v16h bv = *(const v16h*)(Vp + ((((size_t)bh * 32 + j) * 4 + nt) << 9) + lane * 16);
      o[nt] = WMMA_F16(pa, bv, o[nt]);
    }
  }

  // finalize: divide by row sums, scatter as f16 A-fragments for GEMM2
  float inv[8];
#pragma unroll
  for (int r = 0; r < 8; ++r) inv[r] = 1.0f / l8[r];
  int b = bh >> 3, h = bh & 7;
#pragma unroll
  for (int nt = 0; nt < 4; ++nt)
#pragma unroll
    for (int r = 0; r < 8; ++r) {
      float val = o[nt][r] * inv[r];
      int trow  = qb * 16 + r + 8 * half;
      int m     = b * TT + trow;
      int c     = h * DH + nt * 16 + col;
      int tileM = m >> 4;
      int tileK = c >> 5;
      int kk    = c & 31;
      int lp = (trow & 15) + 16 * (((kk & 15) >= 8) ? 1 : 0);
      int v  = (kk < 16) ? ((kk & 7) >> 1) : (4 + ((kk & 7) >> 1));
      int e  = 2 * v + (kk & 1);
      Yp[(((size_t)tileM * KT + tileK) << 9) + lp * 16 + e] = (_Float16)val;
    }
}

// GEMM2: out = y @ Wout, plain f32 store.
__global__ __launch_bounds__(256) void gemm_out(const _Float16* __restrict__ Ap,
                                                const _Float16* __restrict__ Bp,
                                                float* __restrict__ out) {
  int lane = threadIdx.x & 31;
  int w    = threadIdx.x >> 5;
  int tileM0 = blockIdx.y * 8 + (w >> 1) * 2;
  int tileN0 = blockIdx.x * 8 + (w & 1) * 4;
  v8f acc[2][4];
#pragma unroll
  for (int i = 0; i < 2; ++i)
#pragma unroll
    for (int j = 0; j < 4; ++j)
#pragma unroll
      for (int r = 0; r < 8; ++r) acc[i][j][r] = 0.0f;

  gemm_wave_32x64(Ap, Bp, KT, tileM0, tileN0, lane, acc);

  int half = lane >> 4, col = lane & 15;
#pragma unroll
  for (int i = 0; i < 2; ++i)
#pragma unroll
    for (int j = 0; j < 4; ++j)
#pragma unroll
      for (int r = 0; r < 8; ++r) {
        int m = (tileM0 + i) * 16 + r + 8 * half;
        int n = (tileN0 + j) * 16 + col;
        out[(size_t)m * CC + n] = acc[i][j][r];
      }
}

extern "C" void kernel_launch(void* const* d_in, const int* in_sizes, int n_in,
                              void* d_out, int out_size, void* d_ws, size_t ws_size,
                              hipStream_t stream) {
  (void)in_sizes; (void)n_in; (void)out_size; (void)ws_size;
  const float* x    = (const float*)d_in[0];
  const float* Wqkv = (const float*)d_in[1];
  const float* Wout = (const float*)d_in[2];
  float* out = (float*)d_out;

  char* ws = (char*)d_ws;
  size_t off = 0;
  auto take = [&](size_t bytes) {
    char* p = ws + off;
    off += (bytes + 255) & ~(size_t)255;
    return p;
  };
  const size_t halfsX = (size_t)M1 * CC;          // 8.4M halfs
  _Float16* xP    = (_Float16*)take(halfsX * 2);                 // x packed A
  _Float16* WqkvP = (_Float16*)take((size_t)CC * 3 * CC * 2);    // Wqkv packed B
  _Float16* WoutP = (_Float16*)take((size_t)CC * CC * 2);        // Wout packed B
  _Float16* Qp    = (_Float16*)take(halfsX * 2);                 // Q A-fragments
  _Float16* Kp    = (_Float16*)take(halfsX * 2);                 // K B-fragments
  _Float16* Vp    = (_Float16*)take(halfsX * 2);                 // V B-fragments
  _Float16* Yp    = (_Float16*)take(halfsX * 2);                 // attn out A-frag

  // 1. pack activations and weights to f16 fragment tiles
  pack_a_f16<<<(int)(halfsX / 256), 256, 0, stream>>>(x, xP, CC);
  pack_b_f16<<<(CC * 3 * CC) / 256, 256, 0, stream>>>(Wqkv, WqkvP, CC, 3 * CC);
  pack_b_f16<<<(CC * CC) / 256, 256, 0, stream>>>(Wout, WoutP, CC, CC);

  // 2. QKV projection (M=16384, K=512, N=1536), scatter into Q/K/V packs
  gemm_qkv<<<dim3(3 * CC / 128, M1 / 128), 256, 0, stream>>>(xP, WqkvP, Qp, Kp, Vp);

  // 3. causal flash attention: 128 (b,h) x 64 query tiles = 8192 waves
  attn_flash<<<(BB * HH * (TT / 16)) / 8, 256, 0, stream>>>(Qp, Kp, Vp, Yp);

  // 4. output projection (M=16384, K=512, N=512)
  gemm_out<<<dim3(CC / 128, M1 / 128), 256, 0, stream>>>(Yp, WoutP, out);
}
